// CausalAttention_ov3_57518202028182
// MI455X (gfx1250) — matrix operand order repaired and checked by reading, not compiled, not run on hardware
//
#include <hip/hip_runtime.h>
#include <cstdint>
#include <cstddef>

// ---------------------------------------------------------------------------
// Causal attention, N=4096, D=1024, f32 in/out, bf16 WMMA compute (gfx1250).
//
// Pipeline (all on `stream`, workspace layout below):
//   1) cvt      : x, wq, wk, wv  f32 -> bf16 (vectorized)
//   2) qkv      : Q = x wq^T, K = x wk^T, Vt = (x wv^T)^T      [WMMA bf16]
//   3) scores   : S = (Q K^T)/32, causal mask; 16q x 64k per wave,
//                 software-pipelined (loads for d+32 in flight during wmma d)
//   4) softmax  : row softmax over keys 0..row + dropout -> P bf16 (zero-pad
//                 each row to a 32-key boundary so PV never reads garbage)
//   5) out      : O = P V (via Vt); 16t x 64f per wave, software-pipelined
//
// Workspace (needs ~134 MB):
//   [0,8M)    xb      4096x1024 bf16
//   [8M,10M)  wqb     1024x1024 bf16
//   [10M,12M) wkb
//   [12M,14M) wvb
//   [14M,22M) Qb      4096x1024 bf16 row-major
//   [22M,30M) Kb      4096x1024 bf16 row-major
//   [30M,38M) Vtb     1024x4096 bf16 row-major (V transposed)
//   [38M,102M) S      4096x4096 f32
//   [102M,134M) P     4096x4096 bf16
// ---------------------------------------------------------------------------

#define N_TOK 4096
#define DIM   1024
#define DROP_RATE 0.1f

typedef __bf16 bf16;
typedef __attribute__((ext_vector_type(16))) __bf16 v16bf;
typedef __attribute__((ext_vector_type(8)))  __bf16 v8bf;
typedef __attribute__((ext_vector_type(4)))  __bf16 v4bf;
typedef __attribute__((ext_vector_type(8)))  float  v8f;

// ---- WMMA fragment helpers (wave32, ISA 7.12.2 layouts) --------------------

// A-matrix 16x32 bf16: lanes 0-15 hold row=lane, K {0..7, 16..23};
// lanes 16-31 hold row=lane-16, K {8..15, 24..31}.  base -> (tile_row0, k0).
__device__ __forceinline__ v16bf load_a_frag(const bf16* __restrict__ base, int ld) {
  const int lane = threadIdx.x & 31;
  const int row  = lane & 15;
  const int half = lane >> 4;
  const bf16* p  = base + (size_t)row * ld + 8 * half;
  v8bf lo = *(const v8bf*)(p);
  v8bf hi = *(const v8bf*)(p + 16);
  v16bf a;
#pragma unroll
  for (int e = 0; e < 8; ++e) { a[e] = lo[e]; a[e + 8] = hi[e]; }
  return a;
}

// B-matrix 32x16 bf16: lane = output column; lanes 0-15 hold K {0..7, 16..23},
// lanes 16-31 hold K {8..15, 24..31}.  `base` -> row-major stationary matrix whose
// row `col` supplies column `col` of B (i.e. C = A * base^T).
__device__ __forceinline__ v16bf load_b_frag(const bf16* __restrict__ base, int ld) {
  const int lane = threadIdx.x & 31;
  const int col  = lane & 15;
  const int half = lane >> 4;
  const bf16* p  = base + (size_t)col * ld + 8 * half;
  v8bf lo = *(const v8bf*)(p);
  v8bf hi = *(const v8bf*)(p + 16);
  v16bf b;
#pragma unroll
  for (int e = 0; e < 8; ++e) { b[e] = lo[e]; b[e + 8] = hi[e]; }
  return b;
}

__device__ __forceinline__ v8f wmma_bf16(v16bf a, v16bf b, v8f c) {
  return __builtin_amdgcn_wmma_f32_16x16x32_bf16(
      /*neg_a=*/false, a, /*neg_b=*/false, b,
      /*c_mod=*/(short)0, c, /*reuse_a=*/false, /*reuse_b=*/false);
}

// ---- 1) f32 -> bf16 conversion (4 elems/thread) ---------------------------

__global__ __launch_bounds__(256) void cvt_kernel(
    const float* __restrict__ x,  const float* __restrict__ wq,
    const float* __restrict__ wk, const float* __restrict__ wv,
    bf16* __restrict__ xb,  bf16* __restrict__ wqb,
    bf16* __restrict__ wkb, bf16* __restrict__ wvb) {
  const int NX = N_TOK * DIM;   // 4 Mi
  const int NW = DIM * DIM;     // 1 Mi
  int i = (blockIdx.x * 256 + threadIdx.x) * 4;   // region bounds are 4-aligned
  const float* src;
  bf16* dst;
  int o;
  if (i < NX)               { src = x;  dst = xb;  o = i; }
  else if (i < NX + NW)     { src = wq; dst = wqb; o = i - NX; }
  else if (i < NX + 2 * NW) { src = wk; dst = wkb; o = i - NX - NW; }
  else                      { src = wv; dst = wvb; o = i - NX - 2 * NW; }
  float4 v = *(const float4*)(src + o);
  v4bf b;
  b[0] = (bf16)v.x; b[1] = (bf16)v.y; b[2] = (bf16)v.z; b[3] = (bf16)v.w;
  *(v4bf*)(dst + o) = b;
}

// ---- 2) QKV projection: 3 WMMA chains share each A-frag, pipelined ---------

__global__ __launch_bounds__(256) void qkv_kernel(
    const bf16* __restrict__ xb,  const bf16* __restrict__ wqb,
    const bf16* __restrict__ wkb, const bf16* __restrict__ wvb,
    bf16* __restrict__ Qb, bf16* __restrict__ Kb, bf16* __restrict__ Vtb) {
  const int w    = threadIdx.x >> 5;
  const int lane = threadIdx.x & 31;
  const int t = blockIdx.x * 8 + w;       // 16384 tiles
  const int ti = t >> 6;                  // token tile   0..255
  const int tj = t & 63;                  // feature tile 0..63

  const bf16* abase = xb  + (size_t)(ti * 16) * DIM;
  const bf16* qw    = wqb + (size_t)(tj * 16) * DIM;
  const bf16* kw    = wkb + (size_t)(tj * 16) * DIM;
  const bf16* vw    = wvb + (size_t)(tj * 16) * DIM;

  v8f cq = {}, ck = {}, cv = {};
  // prologue: frags for d = 0
  v16bf a_c  = load_a_frag(abase, DIM);
  v16bf bq_c = load_b_frag(qw, DIM);
  v16bf bk_c = load_b_frag(kw, DIM);
  v16bf bv_c = load_b_frag(vw, DIM);
  for (int d = 0; d < DIM - 32; d += 32) {
    const int dn = d + 32;
    // issue next-iteration loads BEFORE consuming current frags
    v16bf a_n  = load_a_frag(abase + dn, DIM);
    v16bf bq_n = load_b_frag(qw + dn, DIM);
    v16bf bk_n = load_b_frag(kw + dn, DIM);
    v16bf bv_n = load_b_frag(vw + dn, DIM);
    cq = wmma_bf16(a_c, bq_c, cq);
    ck = wmma_bf16(a_c, bk_c, ck);
    cv = wmma_bf16(a_c, bv_c, cv);
    a_c = a_n; bq_c = bq_n; bk_c = bk_n; bv_c = bv_n;
  }
  cq = wmma_bf16(a_c, bq_c, cq);
  ck = wmma_bf16(a_c, bk_c, ck);
  cv = wmma_bf16(a_c, bv_c, cv);

  const int ncol  = lane & 15;
  const int mbase = (lane >> 4) * 8;
  const int m0 = ti * 16 + mbase;
  const int f  = tj * 16 + ncol;
#pragma unroll
  for (int r = 0; r < 8; ++r) {
    Qb[(size_t)(m0 + r) * DIM + f] = (bf16)cq[r];
    Kb[(size_t)(m0 + r) * DIM + f] = (bf16)ck[r];
  }
  v8bf vv;
#pragma unroll
  for (int r = 0; r < 8; ++r) vv[r] = (bf16)cv[r];
  *(v8bf*)(Vtb + (size_t)f * N_TOK + m0) = vv;   // transposed store, contiguous
}

// ---- 3) S = (Q K^T)/sqrt(D): 16q x 64k per wave, pipelined -----------------

__global__ __launch_bounds__(256) void scores_kernel(
    const bf16* __restrict__ Qb, const bf16* __restrict__ Kb,
    float* __restrict__ S) {
  const int w    = threadIdx.x >> 5;
  const int lane = threadIdx.x & 31;
  const int ti = blockIdx.y;              // query tile     0..255
  const int kb = blockIdx.x * 8 + w;      // 64-key block   0..63
  if (kb * 4 > ti) return;                // block entirely above diagonal

  const bf16* qbase = Qb + (size_t)(ti * 16) * DIM;
  const bf16* kbase = Kb + (size_t)(kb * 64) * DIM;

  v8f acc[4] = {};
  v16bf a_c = load_a_frag(qbase, DIM);
  v16bf b_c[4];
#pragma unroll
  for (int s = 0; s < 4; ++s) b_c[s] = load_b_frag(kbase + (size_t)(s * 16) * DIM, DIM);

  for (int d = 0; d < DIM - 32; d += 32) {
    const int dn = d + 32;
    v16bf a_n = load_a_frag(qbase + dn, DIM);
    v16bf b_n[4];
#pragma unroll
    for (int s = 0; s < 4; ++s)
      b_n[s] = load_b_frag(kbase + (size_t)(s * 16) * DIM + dn, DIM);
#pragma unroll
    for (int s = 0; s < 4; ++s) acc[s] = wmma_bf16(a_c, b_c[s], acc[s]);
    a_c = a_n;
#pragma unroll
    for (int s = 0; s < 4; ++s) b_c[s] = b_n[s];
  }
#pragma unroll
  for (int s = 0; s < 4; ++s) acc[s] = wmma_bf16(a_c, b_c[s], acc[s]);

  const int ncol  = lane & 15;
  const int mbase = (lane >> 4) * 8;
  const float scale = 0.03125f;           // 1/sqrt(1024)
#pragma unroll
  for (int s = 0; s < 4; ++s) {
    const int key = kb * 64 + s * 16 + ncol;
#pragma unroll
    for (int r = 0; r < 8; ++r) {
      int qrow = ti * 16 + mbase + r;
      float v = acc[s][r] * scale;
      if (key > qrow) v = -__builtin_inff();
      S[(size_t)qrow * N_TOK + key] = v;
    }
  }
}

// ---- 4) row softmax + dropout -> P (bf16, zero-padded to 32-key boundary) --

__device__ __forceinline__ float hash_rand01(uint32_t v) {
  v += 42u;                               // "key"
  v ^= v >> 17; v *= 0xed5ad4bbu;
  v ^= v >> 11; v *= 0xac4c1b51u;
  v ^= v >> 15; v *= 0x31848babu;
  v ^= v >> 14;
  return (float)(v >> 8) * (1.0f / 16777216.0f);
}

__global__ __launch_bounds__(256) void softmax_kernel(
    const float* __restrict__ S, bf16* __restrict__ P) {
  const int row = blockIdx.x;
  const int tid = threadIdx.x;
  const int nvalid = row + 1;
  const int nfill  = (nvalid + 31) & ~31;
  const float* srow = S + (size_t)row * N_TOK;
  bf16*        prow = P + (size_t)row * N_TOK;

  __shared__ float red[8];

  // pass 1: row max
  float m = -__builtin_inff();
  for (int c = tid; c < nvalid; c += 256) m = fmaxf(m, srow[c]);
#pragma unroll
  for (int o = 16; o > 0; o >>= 1) m = fmaxf(m, __shfl_xor(m, o, 32));
  if ((tid & 31) == 0) red[tid >> 5] = m;
  __syncthreads();
  m = red[0];
#pragma unroll
  for (int k = 1; k < 8; ++k) m = fmaxf(m, red[k]);
  __syncthreads();

  // pass 2: sum of exp
  float s = 0.f;
  for (int c = tid; c < nvalid; c += 256) s += __expf(srow[c] - m);
#pragma unroll
  for (int o = 16; o > 0; o >>= 1) s += __shfl_xor(s, o, 32);
  if ((tid & 31) == 0) red[tid >> 5] = s;
  __syncthreads();
  s = red[0];
#pragma unroll
  for (int k = 1; k < 8; ++k) s += red[k];
  const float inv_sum = 1.0f / s;
  const float inv_keep = 1.0f / (1.0f - DROP_RATE);

  // pass 3: normalize + dropout + bf16 store (zeros pad to 32-boundary)
  for (int c = tid; c < nfill; c += 256) {
    float p = 0.f;
    if (c < nvalid) {
      float e = __expf(srow[c] - m) * inv_sum;
      uint32_t idx = (uint32_t)row * (uint32_t)N_TOK + (uint32_t)c;
      p = (hash_rand01(idx) >= DROP_RATE) ? e * inv_keep : 0.f;
    }
    prow[c] = (bf16)p;
  }
}

// ---- 5) O = P @ V: 16t x 64f per wave, pipelined ---------------------------

__global__ __launch_bounds__(256) void out_kernel(
    const bf16* __restrict__ P, const bf16* __restrict__ Vtb,
    float* __restrict__ O) {
  const int w    = threadIdx.x >> 5;
  const int lane = threadIdx.x & 31;
  const int t = blockIdx.x * 8 + w;       // 4096 blocks of work
  const int ti = t >> 4;                  // token tile        0..255
  const int fb = t & 15;                  // 64-feature block  0..15

  const bf16* pbase = P   + (size_t)(ti * 16) * N_TOK;
  const bf16* vbase = Vtb + (size_t)(fb * 64) * N_TOK;
  const int nk = ((ti * 16 + 16) + 31) & ~31;   // keys, rounded to 32

  v8f acc[4] = {};
  v16bf a_c = load_a_frag(pbase, N_TOK);
  v16bf b_c[4];
#pragma unroll
  for (int s = 0; s < 4; ++s)
    b_c[s] = load_b_frag(vbase + (size_t)(s * 16) * N_TOK, N_TOK);

  for (int k = 0; k < nk - 32; k += 32) {
    const int kn = k + 32;
    v16bf a_n = load_a_frag(pbase + kn, N_TOK);
    v16bf b_n[4];
#pragma unroll
    for (int s = 0; s < 4; ++s)
      b_n[s] = load_b_frag(vbase + (size_t)(s * 16) * N_TOK + kn, N_TOK);
#pragma unroll
    for (int s = 0; s < 4; ++s) acc[s] = wmma_bf16(a_c, b_c[s], acc[s]);
    a_c = a_n;
#pragma unroll
    for (int s = 0; s < 4; ++s) b_c[s] = b_n[s];
  }
#pragma unroll
  for (int s = 0; s < 4; ++s) acc[s] = wmma_bf16(a_c, b_c[s], acc[s]);

  const int ncol  = lane & 15;
  const int mbase = (lane >> 4) * 8;
#pragma unroll
  for (int s = 0; s < 4; ++s) {
    const int fo = fb * 64 + s * 16 + ncol;
#pragma unroll
    for (int r = 0; r < 8; ++r)
      O[(size_t)(ti * 16 + mbase + r) * DIM + fo] = acc[s][r];
  }
}

// ---- host launch -----------------------------------------------------------

extern "C" void kernel_launch(void* const* d_in, const int* in_sizes, int n_in,
                              void* d_out, int out_size, void* d_ws, size_t ws_size,
                              hipStream_t stream) {
  const float* x  = (const float*)d_in[0];
  const float* wq = (const float*)d_in[1];
  const float* wk = (const float*)d_in[2];
  const float* wv = (const float*)d_in[3];
  float* out = (float*)d_out;

  char* ws = (char*)d_ws;
  const size_t MB = 1u << 20;
  bf16*  xb  = (bf16*)(ws + 0 * MB);
  bf16*  wqb = (bf16*)(ws + 8 * MB);
  bf16*  wkb = (bf16*)(ws + 10 * MB);
  bf16*  wvb = (bf16*)(ws + 12 * MB);
  bf16*  Qb  = (bf16*)(ws + 14 * MB);
  bf16*  Kb  = (bf16*)(ws + 22 * MB);
  bf16*  Vtb = (bf16*)(ws + 30 * MB);
  float* S   = (float*)(ws + 38 * MB);
  bf16*  P   = (bf16*)(ws + 102 * MB);

  // 1) convert: 7340032 elems / (256 thr * 4 elems)
  cvt_kernel<<<7168, 256, 0, stream>>>(x, wq, wk, wv, xb, wqb, wkb, wvb);
  // 2) QKV: 256*64 tiles, 8 waves/block
  qkv_kernel<<<2048, 256, 0, stream>>>(xb, wqb, wkb, wvb, Qb, Kb, Vtb);
  // 3) scores: grid (8 key-block groups, 256 query tiles), 64 keys per wave
  scores_kernel<<<dim3(8, 256), 256, 0, stream>>>(Qb, Kb, S);
  // 4) softmax+dropout: one block per query row
  softmax_kernel<<<N_TOK, 256, 0, stream>>>(S, P);
  // 5) output GEMM: 4096 wave-blocks / 8 waves
  out_kernel<<<512, 256, 0, stream>>>(P, Vtb, out);
}
